// RosaAttention_40973988004494
// MI455X (gfx1250) — matrix operand order repaired and checked by reading, not compiled, not run on hardware
//
#include <hip/hip_runtime.h>
#include <hip/hip_bf16.h>
#include <math.h>

// RosaAttention constants
#define H_   16
#define KV_  4
#define QKB_ 64
#define VB_  64
#define HID_ 2048
#define B_   2
#define S_   2048

#define ACT_NONE 0
#define ACT_TANH 1
#define ACT_SIG  2

// store modes for GEMM epilogue
#define ST_F32      0   // float, row-major [M][N]
#define ST_BF16     1   // bf16, row-major [M][N]
#define ST_BF16_VT  2   // bf16, v-transposed [b][kv][vb][S]

typedef __attribute__((ext_vector_type(16))) __bf16 v16bf;
typedef __attribute__((ext_vector_type(8)))  __bf16 v8bf;
typedef __attribute__((ext_vector_type(8)))  float  v8f;
typedef __attribute__((ext_vector_type(4)))  unsigned int v4u;
typedef __attribute__((ext_vector_type(8)))  int v8i;
typedef __attribute__((ext_vector_type(4)))  int v4i;

// ---------------------------------------------------------------------------
// WMMA fragment helpers (wave32, 16x16x32 bf16).
// A (16x32, MxK): lane = (hf, r): row r = lane&15; hf = lane>>4 selects
//   K in {hf*8..hf*8+7} and {16+hf*8..16+hf*8+7}  (ISA 7.12.2, 16-bit A).
// B (32x16, KxN): lane = (hf, n): column n = lane&15; hf selects
//   K in {hf*8..hf*8+7} and {16+hf*8..16+hf*8+7} (same order as A).
// C/D (16x16 f32): VGPR j <-> row (hf*8 + j), lane&15 <-> column.
// ---------------------------------------------------------------------------
static __device__ inline v16bf load_frag_a(const __bf16* base, int ld) {
  const int lane = threadIdx.x & 31;
  const int hf   = lane >> 4;
  const __bf16* p = base + (size_t)(lane & 15) * ld + hf * 8;
  v8bf lo = *(const v8bf*)(p);
  v8bf hi = *(const v8bf*)(p + 16);
  v16bf f;
#pragma unroll
  for (int e = 0; e < 8; ++e) { f[e] = lo[e]; f[8 + e] = hi[e]; }
  return f;
}

static __device__ inline v16bf load_frag_b(const __bf16* base, int ld) {
  const int lane = threadIdx.x & 31;
  const int hf   = lane >> 4;
  const __bf16* p = base + (size_t)(lane & 15) * ld + hf * 8;
  v8bf lo = *(const v8bf*)(p);
  v8bf hi = *(const v8bf*)(p + 16);
  v16bf f;
#pragma unroll
  for (int e = 0; e < 8; ++e) { f[e] = lo[e]; f[8 + e] = hi[e]; }
  return f;
}

static __device__ inline v8f wmma_bf16(v16bf a, v16bf b, v8f c) {
  return __builtin_amdgcn_wmma_f32_16x16x32_bf16(
      /*neg_a=*/false, a, /*neg_b=*/false, b,
      /*c_mod=*/(short)0, c, /*reuse_a=*/false, /*reuse_b=*/false);
}

// ---------------------------------------------------------------------------
// Tensor Data Mover: DMA a 2-D bf16 tile (tile1 rows x tile0 elems, row
// stride `stride0` elems) from global into LDS at byte offset `lds_off`,
// packed row-major [tile1][tile0]. D# layout per CDNA5 ISA.
// This toolchain exposes the 6-arg builtin form:
//   (g0 u32x4, g1 i32x8, g2 i32x4, g3 i32x4, g4 i32x8, cpol) — g4 zero here
// since a 2-D tensor needs only descriptor groups 0/1.
// ---------------------------------------------------------------------------
static __device__ inline void tdm_load_2d_bf16(const __bf16* gptr, unsigned lds_off,
                                               int tile0, int tile1, int stride0) {
  unsigned long long ga = (unsigned long long)(uintptr_t)gptr;
  v4u g0;
  g0[0] = 1u;                                   // count=1, user mode, no gather
  g0[1] = lds_off;                              // lds_addr (bytes)
  g0[2] = (unsigned)ga;                         // global_addr[31:0]
  g0[3] = (unsigned)(ga >> 32) | (2u << 30);    // global_addr[56:32] | type=2
  v8i g1;
  g1[0] = (1 << 16);                            // data_size=1 (2 bytes)
  g1[1] = (int)(((unsigned)tile0 & 0xFFFFu) << 16);            // tensor_dim0 lo16
  g1[2] = (int)((((unsigned)tile0 >> 16) & 0xFFFFu) |
                (((unsigned)tile1 & 0xFFFFu) << 16));          // dim0 hi | dim1 lo
  g1[3] = (int)((((unsigned)tile1 >> 16) & 0xFFFFu) |
                (((unsigned)tile0 & 0xFFFFu) << 16));          // dim1 hi | tile_dim0
  g1[4] = (int)((unsigned)tile1 & 0xFFFFu);                    // tile_dim1 (dim2=0)
  g1[5] = stride0;                              // tensor_dim0_stride[31:0] (elems)
  g1[6] = 0;                                    // stride0 hi | stride1 lo
  g1[7] = 0;
  v4i g2 = {0, 0, 0, 0};
  v4i g3 = {0, 0, 0, 0};
  v8i g4 = {0, 0, 0, 0, 0, 0, 0, 0};
  __builtin_amdgcn_tensor_load_to_lds(g0, g1, g2, g3, g4, 0);
}

// ---------------------------------------------------------------------------
// Conversions. Weights are transposed once to [N][K] bf16 so GEMM B-tiles
// are contiguous-K and TDM-friendly.
// ---------------------------------------------------------------------------
__global__ void cvt_f32_to_bf16(const float* __restrict__ src,
                                __bf16* __restrict__ dst, int n) {
  int i = blockIdx.x * blockDim.x + threadIdx.x;
  if (i < n) dst[i] = (__bf16)src[i];
}

// dst[n][k] = (bf16) src[k][n];  src is [K][N] fp32
__global__ void cvt_transpose_bf16(const float* __restrict__ src,
                                   __bf16* __restrict__ dst, int N, int K) {
  int i = blockIdx.x * blockDim.x + threadIdx.x;
  if (i >= N * K) return;
  int n = i / K, k = i - n * K;
  dst[i] = (__bf16)src[(size_t)k * N + n];
}

// bias[n] = sum_k emb0_flat[k] * Wo[k, n]   (K = H*VB = 1024, N = HID)
__global__ void bias_kernel(const float* __restrict__ emb0,
                            const float* __restrict__ Wo,
                            float* __restrict__ bias) {
  int n = blockIdx.x * blockDim.x + threadIdx.x;
  if (n >= HID_) return;
  float s = 0.f;
  for (int k = 0; k < H_ * VB_; ++k) s += emb0[k] * Wo[(size_t)k * HID_ + n];
  bias[n] = s;
}

// ---------------------------------------------------------------------------
// Tiled bf16 WMMA GEMM: Out = act(A[M][K] @ Bt[N][K]^T) (+ bias[n]).
// Block tile 128x64, BK=64, 256 threads = 8 wave32 waves (4x2 wave grid,
// 32x32 per wave). Double-buffered LDS fed by the Tensor Data Mover:
// wave 0 issues the next K-slab's two TDM descriptors while all waves run
// WMMAs on the current buffer; TENSORcnt gates buffer reuse.
// ---------------------------------------------------------------------------
__global__ __launch_bounds__(256)
void gemm_bf16_tn_kernel(const __bf16* __restrict__ A,   // [M][K]
                         const __bf16* __restrict__ Bt,  // [N][K]
                         void* __restrict__ Out, const float* __restrict__ bias,
                         int M, int N, int K, int act, int store_mode) {
  __shared__ __attribute__((aligned(16))) __bf16 sA[2][128 * 64];  // [m][k]
  __shared__ __attribute__((aligned(16))) __bf16 sB[2][64 * 64];   // [n][k]
  const int tid  = threadIdx.x;
  const int wave = tid >> 5;
  const int wm   = wave >> 1;   // 0..3
  const int wn   = wave & 1;    // 0..1
  const int lane = tid & 31;
  const int hf   = lane >> 4;
  const int n15  = lane & 15;
  const int m0   = blockIdx.y * 128;
  const int nb0  = blockIdx.x * 64;
  const int wave0 = (wave == 0);

  v8f acc[4] = {};

  const unsigned ldsA0 = (unsigned)(uintptr_t)&sA[0][0];
  const unsigned ldsA1 = (unsigned)(uintptr_t)&sA[1][0];
  const unsigned ldsB0 = (unsigned)(uintptr_t)&sB[0][0];
  const unsigned ldsB1 = (unsigned)(uintptr_t)&sB[1][0];

  const int nstage = K >> 6;
  if (wave0) {
    tdm_load_2d_bf16(A  + (size_t)m0  * K, ldsA0, 64, 128, K);
    tdm_load_2d_bf16(Bt + (size_t)nb0 * K, ldsB0, 64, 64,  K);
  }

  for (int it = 0; it < nstage; ++it) {
    const int kk  = it << 6;
    const int cur = it & 1;
    if (wave0) {
      if (it + 1 < nstage) {
        // issue next slab into the other buffer, then wait for current pair
        tdm_load_2d_bf16(A  + (size_t)m0  * K + kk + 64, cur ? ldsA0 : ldsA1,
                         64, 128, K);
        tdm_load_2d_bf16(Bt + (size_t)nb0 * K + kk + 64, cur ? ldsB0 : ldsB1,
                         64, 64, K);
        __builtin_amdgcn_s_wait_tensorcnt(2);
      } else {
        __builtin_amdgcn_s_wait_tensorcnt(0);
      }
    }
    __syncthreads();   // current buffer visible to all waves

    const __bf16* cA = sA[cur];
    const __bf16* cB = sB[cur];
#pragma unroll
    for (int ks = 0; ks < 2; ++ks) {
      v16bf a0 = load_frag_a(cA + (wm * 32)      * 64 + ks * 32, 64);
      v16bf a1 = load_frag_a(cA + (wm * 32 + 16) * 64 + ks * 32, 64);
      v16bf b0 = load_frag_b(cB + (wn * 32)      * 64 + ks * 32, 64);
      v16bf b1 = load_frag_b(cB + (wn * 32 + 16) * 64 + ks * 32, 64);
      acc[0] = wmma_bf16(a0, b0, acc[0]);
      acc[1] = wmma_bf16(a0, b1, acc[1]);
      acc[2] = wmma_bf16(a1, b0, acc[2]);
      acc[3] = wmma_bf16(a1, b1, acc[3]);
    }
    __syncthreads();   // all reads done before this buffer is re-targeted
  }

  // Epilogue: activation (+bias), store per mode.
#pragma unroll
  for (int mi = 0; mi < 2; ++mi)
#pragma unroll
    for (int ni = 0; ni < 2; ++ni) {
      v8f a = acc[mi * 2 + ni];
#pragma unroll
      for (int j = 0; j < 8; ++j) {
        int row = m0 + wm * 32 + mi * 16 + hf * 8 + j;
        int col = nb0 + wn * 32 + ni * 16 + n15;
        float v = a[j];
        if (act == ACT_TANH)      v = tanhf(v);
        else if (act == ACT_SIG)  v = 1.f / (1.f + __expf(-v));
        if (bias) v += bias[col];
        if (store_mode == ST_F32) {
          ((float*)Out)[(size_t)row * N + col] = v;
        } else if (store_mode == ST_BF16) {
          ((__bf16*)Out)[(size_t)row * N + col] = (__bf16)v;
        } else {  // ST_BF16_VT: [b][kv][vb][S], row = b*S+s, col = kv*VB+vb
          int bb = row >> 11, s = row & (S_ - 1);
          int kv = col >> 6,  vb = col & 63;
          ((__bf16*)Out)[((size_t)(bb * KV_ + kv) * VB_ + vb) * S_ + s] = (__bf16)v;
        }
      }
    }
}

// ---------------------------------------------------------------------------
// Flash-style causal attention on soft-bit q/k/v (bf16, post-activation).
// Grid: (S/128, H, B). 256 threads = 8 fully independent waves; wave w owns
// 16 q-rows. k frags and v^T frags load straight from global (both are
// K-contiguous per lane), so the loop has NO barriers; only the probs
// C->A re-fragmentation round-trips through per-wave LDS.
// ---------------------------------------------------------------------------
__global__ __launch_bounds__(256)
void rosa_attn_kernel(const __bf16* __restrict__ qb, const __bf16* __restrict__ kb,
                      const __bf16* __restrict__ vT, const int* __restrict__ amask,
                      const float* __restrict__ emb0, const float* __restrict__ emb1,
                      __bf16* __restrict__ ctxp) {
  __shared__ __attribute__((aligned(16))) __bf16 sP[8 * 16 * 64];  // probs per wave
  const int tid  = threadIdx.x;
  const int w    = tid >> 5;
  const int lane = tid & 31;
  const int hf   = lane >> 4;
  const int n15  = lane & 15;
  const int h    = blockIdx.y;
  const int b    = blockIdx.z;
  const int kvh  = h >> 2;                 // repeat_kv: rep = H/KV = 4
  const int q0   = blockIdx.x * 128 + w * 16;
  __bf16* sPw = sP + w * (16 * 64);

  const float NEG = -__builtin_inff();

  // q fragments (two K-steps of 32 over QKB=64), kept in registers.
  const __bf16* qbase = qb + ((size_t)(b * S_ + q0) * H_ + h) * QKB_;
  v16bf qa0 = load_frag_a(qbase + 0,  H_ * QKB_);
  v16bf qa1 = load_frag_a(qbase + 32, H_ * QKB_);

  float mrow[8], lrow[8];
  v8f o[4] = {};
#pragma unroll
  for (int j = 0; j < 8; ++j) { mrow[j] = NEG; lrow[j] = 0.f; }

  float dsc[4];
#pragma unroll
  for (int nv = 0; nv < 4; ++nv) {
    int vb = nv * 16 + n15;
    dsc[nv] = emb1[h * VB_ + vb] - emb0[h * VB_ + vb];
  }

  // v^T base for this (b, kv head): [vb][S], key-contiguous rows.
  const __bf16* vbase = vT + (size_t)(b * KV_ + kvh) * VB_ * S_;

  const int kend = q0 + 16;               // per-wave causal bound (keys <= q0+15)
  for (int kt = 0; kt < kend; kt += 64) {
    // scores: 16(q) x 64(key), K=QKB=64 in two WMMA steps; k frags from global
    // (QKB contiguous per key, stride KV*QKB between keys).
    v8f sc[4];
#pragma unroll
    for (int nt = 0; nt < 4; ++nt) {
      const __bf16* kbase = kb + ((size_t)(b * S_ + kt + nt * 16) * KV_ + kvh) * QKB_;
      v8f c = {};
      c = wmma_bf16(qa0, load_frag_b(kbase + 0,  KV_ * QKB_), c);
      c = wmma_bf16(qa1, load_frag_b(kbase + 32, KV_ * QKB_), c);
      sc[nt] = c;
    }
    // scale + causal/padding mask
#pragma unroll
    for (int nt = 0; nt < 4; ++nt) {
      int kcol = kt + nt * 16 + n15;
      int mk = amask[b * S_ + kcol];
#pragma unroll
      for (int j = 0; j < 8; ++j) {
        int qrow = q0 + hf * 8 + j;
        float s = sc[nt][j] * (1.0f / 64.0f);
        sc[nt][j] = (kcol <= qrow && mk != 0) ? s : NEG;
      }
    }
    // row max (across 4 n-tiles, then across the 16 lanes of this half)
    float tmax[8];
#pragma unroll
    for (int j = 0; j < 8; ++j) {
      float v = fmaxf(fmaxf(sc[0][j], sc[1][j]), fmaxf(sc[2][j], sc[3][j]));
      v = fmaxf(v, __shfl_xor(v, 1, 32));
      v = fmaxf(v, __shfl_xor(v, 2, 32));
      v = fmaxf(v, __shfl_xor(v, 4, 32));
      v = fmaxf(v, __shfl_xor(v, 8, 32));
      tmax[j] = v;
    }
    float mnew[8], corr[8], psum[8];
#pragma unroll
    for (int j = 0; j < 8; ++j) {
      mnew[j] = fmaxf(mrow[j], tmax[j]);
      corr[j] = (mnew[j] == NEG) ? 1.f : __expf(mrow[j] - mnew[j]);
      psum[j] = 0.f;
    }
    // probs; spill to per-wave LDS (bf16) to re-fragment as WMMA-A.
#pragma unroll
    for (int nt = 0; nt < 4; ++nt)
#pragma unroll
      for (int j = 0; j < 8; ++j) {
        float pv = (mnew[j] == NEG) ? 0.f : __expf(sc[nt][j] - mnew[j]);
        psum[j] += pv;
        sPw[(hf * 8 + j) * 64 + nt * 16 + n15] = (__bf16)pv;
      }
#pragma unroll
    for (int j = 0; j < 8; ++j) {
      float v = psum[j];
      v += __shfl_xor(v, 1, 32);
      v += __shfl_xor(v, 2, 32);
      v += __shfl_xor(v, 4, 32);
      v += __shfl_xor(v, 8, 32);
      lrow[j] = lrow[j] * corr[j] + v;
      mrow[j] = mnew[j];
    }
#pragma unroll
    for (int nv = 0; nv < 4; ++nv)
#pragma unroll
      for (int j = 0; j < 8; ++j) o[nv][j] *= corr[j];

    // ctx += P(16x64) @ vp(64x64): K = keys = 64 in two WMMA steps.
    // v^T rows are key-contiguous -> direct global B-fragments.
#pragma unroll
    for (int ks = 0; ks < 2; ++ks) {
      v16bf pa = load_frag_a(sPw + ks * 32, 64);
#pragma unroll
      for (int nv = 0; nv < 4; ++nv) {
        v16bf bv = load_frag_b(vbase + (size_t)(nv * 16) * S_ + kt + ks * 32, S_);
        o[nv] = wmma_bf16(pa, bv, o[nv]);
      }
    }
  }

  // Normalize and fold in per-bit embedding delta; store ctx' bf16.
#pragma unroll
  for (int j = 0; j < 8; ++j) {
    float rinv = (lrow[j] > 0.f) ? 1.f / lrow[j] : 0.f;
    int qrow = q0 + hf * 8 + j;
#pragma unroll
    for (int nv = 0; nv < 4; ++nv) {
      float v = o[nv][j] * rinv * dsc[nv];
      ctxp[((size_t)(b * S_ + qrow) * H_ + h) * VB_ + nv * 16 + n15] = (__bf16)v;
    }
  }
}

// ---------------------------------------------------------------------------
extern "C" void kernel_launch(void* const* d_in, const int* in_sizes, int n_in,
                              void* d_out, int out_size, void* d_ws, size_t ws_size,
                              hipStream_t stream) {
  const float* x    = (const float*)d_in[0];
  const int*   am   = (const int*)d_in[1];
  const float* Wq   = (const float*)d_in[2];
  const float* Wk   = (const float*)d_in[3];
  const float* Wv   = (const float*)d_in[4];
  const float* Wo   = (const float*)d_in[5];
  const float* emb0 = (const float*)d_in[6];
  const float* emb1 = (const float*)d_in[7];
  float* out = (float*)d_out;
  (void)in_sizes; (void)n_in; (void)out_size; (void)ws_size;

  char* ws = (char*)d_ws;
  size_t off = 0;
  auto take = [&](size_t bytes) -> char* {
    char* p = ws + off;
    off += (bytes + 255) & ~(size_t)255;
    return p;
  };
  const size_t M = (size_t)B_ * S_;                         // 4096 rows
  __bf16* xb   = (__bf16*)take(M * HID_ * 2);               // x (bf16)
  __bf16* wqt  = (__bf16*)take((size_t)(H_ * QKB_) * HID_ * 2);   // Wq^T [N][K]
  __bf16* wkt  = (__bf16*)take((size_t)(KV_ * QKB_) * HID_ * 2);  // Wk^T
  __bf16* wvt  = (__bf16*)take((size_t)(KV_ * VB_) * HID_ * 2);   // Wv^T
  __bf16* wot  = (__bf16*)take((size_t)HID_ * (H_ * VB_) * 2);    // Wo^T
  __bf16* qbb  = (__bf16*)take(M * H_ * QKB_ * 2);          // tanh(x@Wq)
  __bf16* kbb  = (__bf16*)take(M * KV_ * QKB_ * 2);         // tanh(x@Wk)
  __bf16* vTb  = (__bf16*)take((size_t)B_ * KV_ * VB_ * S_ * 2);  // sigmoid(x@Wv)^T
  __bf16* ctx  = (__bf16*)take(M * H_ * VB_ * 2);           // ctx*(emb1-emb0)
  float*  bias = (float*)take(HID_ * sizeof(float));        // emb0 @ Wo

  cvt_f32_to_bf16<<<(unsigned)((M * HID_ + 255) / 256), 256, 0, stream>>>(
      x, xb, (int)(M * HID_));
  auto cvtT = [&](const float* s, __bf16* d, int N, int K) {
    cvt_transpose_bf16<<<(unsigned)(((size_t)N * K + 255) / 256), 256, 0, stream>>>(
        s, d, N, K);
  };
  cvtT(Wq, wqt, H_ * QKB_,  HID_);
  cvtT(Wk, wkt, KV_ * QKB_, HID_);
  cvtT(Wv, wvt, KV_ * VB_,  HID_);
  cvtT(Wo, wot, HID_,       H_ * VB_);
  bias_kernel<<<HID_ / 256, 256, 0, stream>>>(emb0, Wo, bias);

  // QKV projections with fused activations (bf16 WMMA, f32 accumulate, TDM).
  gemm_bf16_tn_kernel<<<dim3((H_ * QKB_) / 64, (unsigned)(M / 128)), 256, 0, stream>>>(
      xb, wqt, qbb, nullptr, (int)M, H_ * QKB_, HID_, ACT_TANH, ST_BF16);
  gemm_bf16_tn_kernel<<<dim3((KV_ * QKB_) / 64, (unsigned)(M / 128)), 256, 0, stream>>>(
      xb, wkt, kbb, nullptr, (int)M, KV_ * QKB_, HID_, ACT_TANH, ST_BF16);
  gemm_bf16_tn_kernel<<<dim3((KV_ * VB_) / 64, (unsigned)(M / 128)), 256, 0, stream>>>(
      xb, wvt, vTb, nullptr, (int)M, KV_ * VB_, HID_, ACT_SIG, ST_BF16_VT);

  // Causal attention (flash-style, WMMA for QK^T and PV, barrier-free loop).
  rosa_attn_kernel<<<dim3(S_ / 128, H_, B_), 256, 0, stream>>>(
      qbb, kbb, vTb, am, emb0, emb1, ctx);

  // Output projection: out = ctx' @ Wo + (emb0 @ Wo), f32 result.
  gemm_bf16_tn_kernel<<<dim3(HID_ / 64, (unsigned)(M / 128)), 256, 0, stream>>>(
      ctx, wot, out, bias, (int)M, HID_, H_ * VB_, ACT_NONE, ST_F32);
}
